// GANet_34359738502
// MI455X (gfx1250) — compile-verified, matrix operand order checked by reading, not run on hardware
//
#include <hip/hip_runtime.h>
#include <hip/hip_fp16.h>
#include <stdint.h>

typedef __attribute__((ext_vector_type(16))) _Float16 v16h;
typedef __attribute__((ext_vector_type(8)))  float    v8f;
typedef __attribute__((ext_vector_type(4)))  unsigned int u32x4;
typedef __attribute__((ext_vector_type(4)))  int      i32x4;
typedef __attribute__((ext_vector_type(8)))  int      i32x8;

#define C_ 32
#define D_ 48
#define H_ 64
#define W_ 128
#define HW_ (H_*W_)

#define BFRAG_DW 13824          // 54 fragments * 32 lanes * 8 dwords = 55296 B

// ---------------------------------------------------------------------------
// 1) Transpose x [C,D,H,W] -> xt [C,H,W,D]  (one block per (c,h), LDS staged)
// ---------------------------------------------------------------------------
__global__ void transpose_x_kernel(const float* __restrict__ x, float* __restrict__ xt)
{
    __shared__ float tile[D_ * W_];           // 24 KB
    int c = blockIdx.x >> 6;
    int h = blockIdx.x & 63;
    for (int i = threadIdx.x; i < D_ * W_; i += blockDim.x) {
        int d = i >> 7, w = i & 127;          // coalesced read: lanes -> consecutive w
        tile[i] = x[((size_t)(c * D_ + d) * H_ + h) * W_ + w];
    }
    __syncthreads();
    int w = threadIdx.x;                       // blockDim == 128
    float4* o4 = (float4*)(xt + ((size_t)(c * H_ + h) * W_ + w) * D_);
#pragma unroll
    for (int q = 0; q < 12; ++q) {
        float4 v;
        v.x = tile[(4 * q + 0) * W_ + w];
        v.y = tile[(4 * q + 1) * W_ + w];
        v.z = tile[(4 * q + 2) * W_ + w];
        v.w = tile[(4 * q + 3) * W_ + w];
        o4[q] = v;
    }
}

// ---------------------------------------------------------------------------
// 2) L1-normalize guidance + transpose: g [4,C,5,H,W] -> kt [4,C,H,W,8]
// ---------------------------------------------------------------------------
__global__ void norm_guidance_kernel(const float* __restrict__ g, float* __restrict__ kt)
{
    int t = blockIdx.x * blockDim.x + threadIdx.x;
    if (t >= 4 * C_ * HW_) return;
    int hw = t % HW_;
    int dc = t / HW_;                          // dir*C + c
    float v[5]; float s = 0.f;
#pragma unroll
    for (int i = 0; i < 5; ++i) {
        v[i] = g[((size_t)(dc * 5) + i) * HW_ + hw];
        s += fabsf(v[i]);
    }
    float inv = 1.0f / fmaxf(s, 1e-12f);
    float* o = kt + (size_t)t * 8;
#pragma unroll
    for (int i = 0; i < 5; ++i) o[i] = v[i] * inv;
    o[5] = 0.f; o[6] = 0.f; o[7] = 0.f;
}

// ---------------------------------------------------------------------------
// 3) SGA directional scan, 4-way lane-split over D.
//    Lane quad (sub = lane&3) holds 12 of the 48 d-values each; boundary
//    d+-1 values and the max-over-D are exchanged with wave32 shuffles.
//    AXIS 0: scan along W (rows = (c,h));  AXIS 1: scan along H (rows = (c,w))
//    FIRSTD: write agg;  else: max-combine (RMW) into agg.
// ---------------------------------------------------------------------------
template<int AXIS, int REVD, int FIRSTD>
__global__ void sga_scan_kernel(const float* __restrict__ xt,
                                const float* __restrict__ kt,
                                float* __restrict__ agg, int dir)
{
    int t = blockIdx.x * blockDim.x + threadIdx.x;
    int r   = t >> 2;                          // row
    int sub = t & 3;                           // position within lane quad
    const int nrows = (AXIS == 0) ? C_ * H_ : C_ * W_;
    if (r >= nrows) return;
    int c = 0, h = 0, w = 0;
    if (AXIS == 0) { c = r / H_; h = r % H_; }
    else           { c = r / W_; w = r % W_; }
    const int steps = (AXIS == 0) ? W_ : H_;
    const int stepElem = ((AXIS == 0) ? 1 : W_) * D_ * (REVD ? -1 : 1);

    float A[12];
    for (int s = 0; s < steps; ++s) {
        int pos = REVD ? (steps - 1 - s) : s;
        if (AXIS == 0) w = pos; else h = pos;
        int cbase = ((c * H_ + h) * W_ + w) * D_ + sub * 12;
        const float4* x4 = (const float4*)(xt + cbase);
        if (s + 1 < steps)
            __builtin_prefetch(xt + cbase + stepElem, 0, 0);   // global_prefetch_b8

        if (s == 0) {
#pragma unroll
            for (int q = 0; q < 3; ++q) {
                float4 v = x4[q];
                A[4*q] = v.x; A[4*q+1] = v.y; A[4*q+2] = v.z; A[4*q+3] = v.w;
            }
        } else {
            const float* kw_ = kt + ((size_t)(dir * C_ + c) * HW_ + h * W_ + w) * 8;
            float w0 = kw_[0], w1 = kw_[1], w2 = kw_[2], w3 = kw_[3], w4 = kw_[4];
            // stage x chunk
            float xv[12];
#pragma unroll
            for (int q = 0; q < 3; ++q) {
                float4 v = x4[q];
                xv[4*q] = v.x; xv[4*q+1] = v.y; xv[4*q+2] = v.z; xv[4*q+3] = v.w;
            }
            // max over all 48 d (local 12 + quad combine)
            float mx = A[0];
#pragma unroll
            for (int j = 1; j < 12; ++j) mx = fmaxf(mx, A[j]);
            mx = fmaxf(mx, __shfl_xor(mx, 1, 32));
            mx = fmaxf(mx, __shfl_xor(mx, 2, 32));
            // boundary exchange (old values)
            float prevb = __shfl_up(A[11], 1, 32);
            float nxtb  = __shfl_down(A[0], 1, 32);
            if (sub == 0) prevb = 0.f;
            if (sub == 3) nxtb  = 0.f;
            float prev = prevb;
#pragma unroll
            for (int j = 0; j < 12; ++j) {
                float cur = A[j];
                float nxt = (j < 11) ? A[j + 1] : nxtb;
                A[j] = w0 * xv[j] + w1 * cur + w2 * prev + w3 * nxt + w4 * mx;
                prev = cur;
            }
        }

        float4* o4 = (float4*)(agg + cbase);
        if (FIRSTD) {
#pragma unroll
            for (int q = 0; q < 3; ++q) {
                float4 v; v.x = A[4*q]; v.y = A[4*q+1]; v.z = A[4*q+2]; v.w = A[4*q+3];
                o4[q] = v;
            }
        } else {
#pragma unroll
            for (int q = 0; q < 3; ++q) {
                float4 gv = o4[q];
                float4 v;
                v.x = fmaxf(gv.x, A[4*q]);   v.y = fmaxf(gv.y, A[4*q+1]);
                v.z = fmaxf(gv.z, A[4*q+2]); v.w = fmaxf(gv.w, A[4*q+3]);
                o4[q] = v;
            }
        }
    }
}

// ---------------------------------------------------------------------------
// 4) BN1 + ReLU + convert to f16, layout agg [C,H,W,D] -> yb [D,H,W,C]
//    One wave per (h,w), lane = channel.
// ---------------------------------------------------------------------------
__global__ void bn1_pack_kernel(const float* __restrict__ agg,
                                const float* __restrict__ s1,
                                const float* __restrict__ b1,
                                _Float16* __restrict__ yb)
{
    int t = blockIdx.x * blockDim.x + threadIdx.x;
    int wid = t >> 5, lane = t & 31;
    if (wid >= HW_) return;
    int h = wid / W_, w = wid % W_;
    int c = lane;
    float inv  = s1[c] * rsqrtf(1.0f + 1e-5f);
    float bias = b1[c];
    const float4* a4 = (const float4*)(agg + ((size_t)(c * H_ + h) * W_ + w) * D_);
#pragma unroll
    for (int q = 0; q < 12; ++q) {
        float4 v = a4[q];
        float vv[4] = { v.x, v.y, v.z, v.w };
#pragma unroll
        for (int j = 0; j < 4; ++j) {
            int d = 4 * q + j;
            float y = fmaxf(vv[j] * inv + bias, 0.f);
            yb[((size_t)(d * H_ + h) * W_ + w) * C_ + c] = (_Float16)y;
        }
    }
}

// ---------------------------------------------------------------------------
// 5) Pre-pack weight B-fragments per the 16-bit 32x16 WMMA lane layout.
//    bfrag[tap*2+nt][lane][8 dwords], tap = kd*9+kh*3+kw, nt = out-ch half.
// ---------------------------------------------------------------------------
__global__ void prep_bfrag_kernel(const float* __restrict__ wr, uint32_t* __restrict__ bfrag)
{
    int tapnt = blockIdx.x;                   // 0..53
    int tap = tapnt >> 1, nt = tapnt & 1;
    int lane = threadIdx.x;                   // 32 lanes
    int n = nt * 16 + (lane & 15);
    int khi = (lane >> 4) * 8;
    uint32_t outv[8];
#pragma unroll
    for (int v = 0; v < 8; ++v) {
        int kb = (v < 4) ? (khi + 2 * v) : (16 + khi + 2 * (v - 4));
        float f0 = wr[(n * 32 + kb    ) * 27 + tap];
        float f1 = wr[(n * 32 + kb + 1) * 27 + tap];
        union { _Float16 hh[2]; uint32_t u; } pk;
        pk.hh[0] = (_Float16)f0; pk.hh[1] = (_Float16)f1;
        outv[v] = pk.u;
    }
    uint32_t* dst = bfrag + (size_t)(tapnt * 32 + lane) * 8;
#pragma unroll
    for (int v = 0; v < 8; ++v) dst[v] = outv[v];
}

// ---------------------------------------------------------------------------
// 6) Conv 3x3x3 implicit GEMM with v_wmma_f32_16x16x32_f16.
//    One block per (d,h): 8 waves, wave wt handles the 16-wide w-tile wt.
//    The 55 KB B-fragment table is DMA'd into LDS once per block by the
//    Tensor Data Mover (tensor_load_to_lds + s_wait_tensorcnt), then shared
//    by all 8 waves. LDS region is reused for the epilogue transpose tiles.
// ---------------------------------------------------------------------------
union Frag { uint4 q[2]; v16h v; };

__global__ void conv_wmma_kernel(const _Float16* __restrict__ yb,
                                 const uint32_t* __restrict__ bfrag_g,
                                 const float* __restrict__ x,
                                 const float* __restrict__ s2,
                                 const float* __restrict__ b2,
                                 float* __restrict__ out)
{
    __shared__ __align__(16) uint32_t shbuf[BFRAG_DW];   // 55296 B
    int b  = blockIdx.x;                       // 0 .. D_*H_-1
    int h  = b & 63;
    int d  = b >> 6;
    int wt   = threadIdx.x >> 5;               // wave id = w-tile
    int lane = threadIdx.x & 31;
    int m    = lane & 15;
    int koff = (lane & 16) ? 8 : 0;            // A-fragment K layout per lane half

    // --- TDM: bulk-load the B-fragment table into LDS (wave 0 issues) ---
    if (wt == 0) {
        uint32_t ldsoff = (uint32_t)(uintptr_t)(&shbuf[0]);   // LDS aperture: addr[31:0]
        uint64_t ga = (uint64_t)(uintptr_t)bfrag_g;
        u32x4 g0;
        g0[0] = 1u;                                            // count=1, user mode
        g0[1] = ldsoff;                                        // lds_addr
        g0[2] = (uint32_t)ga;                                  // global_addr[31:0]
        g0[3] = (uint32_t)((ga >> 32) & 0x01FFFFFFu) | (2u << 30); // addr[56:32] | type=2
        i32x8 g1;
        g1[0] = (int)(2u << 16);                // workgroup_mask=0, data_size=4B
        g1[1] = (int)((BFRAG_DW & 0xFFFFu) << 16);   // tensor_dim0[15:0]
        g1[2] = (int)((BFRAG_DW >> 16) | (1u << 16)); // tensor_dim0[31:16] | tensor_dim1=1
        g1[3] = (int)((BFRAG_DW & 0xFFFFu) << 16);   // tensor_dim1 hi=0 | tile_dim0
        g1[4] = 1;                               // tile_dim1=1, tile_dim2=0
        g1[5] = BFRAG_DW;                        // tensor_dim0_stride[31:0]
        g1[6] = 0;
        g1[7] = 0;
        i32x4 z4 = {0, 0, 0, 0};
#if defined(__clang_major__) && (__clang_major__ >= 23)
        i32x8 z8 = {0, 0, 0, 0, 0, 0, 0, 0};
        __builtin_amdgcn_tensor_load_to_lds(g0, g1, z4, z4, z8, 0);
#else
        __builtin_amdgcn_tensor_load_to_lds(g0, g1, z4, z4, 0);
#endif
        __builtin_amdgcn_s_wait_tensorcnt(0);
    }
    __syncthreads();

    v8f c0 = {}; v8f c1 = {};
#pragma unroll
    for (int kd = 0; kd < 3; ++kd) {
        int dp = d + kd - 1;
        if (dp < 0 || dp >= D_) continue;
#pragma unroll
        for (int kh = 0; kh < 3; ++kh) {
            int hp = h + kh - 1;
            if (hp < 0 || hp >= H_) continue;
#pragma unroll
            for (int kw = 0; kw < 3; ++kw) {
                int wp = wt * 16 + m + kw - 1;
                Frag af;
                if (wp >= 0 && wp < W_) {
                    const _Float16* p = yb + ((size_t)(dp * H_ + hp) * W_ + wp) * C_ + koff;
                    af.q[0] = *(const uint4*)p;          // K = koff .. koff+7
                    af.q[1] = *(const uint4*)(p + 16);   // K = koff+16 .. koff+23
                } else {
                    af.q[0] = make_uint4(0, 0, 0, 0);
                    af.q[1] = make_uint4(0, 0, 0, 0);
                }
                int tap = (kd * 3 + kh) * 3 + kw;
                const uint4* bp0 = (const uint4*)(shbuf + (size_t)((tap * 2 + 0) * 32 + lane) * 8);
                const uint4* bp1 = (const uint4*)(shbuf + (size_t)((tap * 2 + 1) * 32 + lane) * 8);
                Frag bf0, bf1;
                bf0.q[0] = bp0[0]; bf0.q[1] = bp0[1];
                bf1.q[0] = bp1[0]; bf1.q[1] = bp1[1];
                c0 = __builtin_amdgcn_wmma_f32_16x16x32_f16(false, af.v, false, bf0.v,
                                                            (short)0, c0, false, false);
                c1 = __builtin_amdgcn_wmma_f32_16x16x32_f16(false, af.v, false, bf1.v,
                                                            (short)0, c1, false, false);
            }
        }
    }

    __syncthreads();   // all waves done reading bfrag LDS; reuse it for tiles

    // D-fragment: VGPR r, lanes 0-15 -> M=r, lanes 16-31 -> M=8+r; N = lane&15.
    float* tile = (float*)shbuf + wt * 512;    // 2 KB per wave
    int mrow = 8 * (lane >> 4);
#pragma unroll
    for (int r = 0; r < 8; ++r) tile[(lane & 15)        * 16 + mrow + r] = c0[r];
#pragma unroll
    for (int r = 0; r < 8; ++r) tile[(16 + (lane & 15)) * 16 + mrow + r] = c1[r];
    // LDS ops from the same wave complete in order; reads below see the writes.

    int n = lane;                              // output channel
    float inv  = s2[n] * rsqrtf(1.0f + 1e-5f);
    float bias = b2[n];
    const float4* t4 = (const float4*)(tile + n * 16);
    size_t obase = ((size_t)(n * D_ + d) * H_ + h) * W_ + wt * 16;
    const float4* x4 = (const float4*)(x + obase);
    float4* o4 = (float4*)(out + obase);
#pragma unroll
    for (int q = 0; q < 4; ++q) {
        float4 v = t4[q];
        float4 rx = x4[q];
        float4 o;
        o.x = fmaxf(v.x * inv + bias + rx.x, 0.f);
        o.y = fmaxf(v.y * inv + bias + rx.y, 0.f);
        o.z = fmaxf(v.z * inv + bias + rx.z, 0.f);
        o.w = fmaxf(v.w * inv + bias + rx.w, 0.f);
        o4[q] = o;
    }
}

// ---------------------------------------------------------------------------
extern "C" void kernel_launch(void* const* d_in, const int* in_sizes, int n_in,
                              void* d_out, int out_size, void* d_ws, size_t ws_size,
                              hipStream_t stream)
{
    const float* x  = (const float*)d_in[0];
    const float* g  = (const float*)d_in[1];
    const float* wr = (const float*)d_in[2];
    const float* s1 = (const float*)d_in[3];
    const float* b1 = (const float*)d_in[4];
    const float* s2 = (const float*)d_in[5];
    const float* b2 = (const float*)d_in[6];
    float* out = (float*)d_out;

    char* ws = (char*)d_ws;
    size_t off = 0;
    auto alloc = [&](size_t bytes) -> void* {
        void* p = ws + off;
        off = (off + bytes + 255) & ~(size_t)255;
        return p;
    };
    const size_t NELEM = (size_t)C_ * D_ * H_ * W_;            // 12,582,912
    float*     xt    = (float*)alloc(NELEM * 4);               // [C,H,W,D]
    float*     kt    = (float*)alloc((size_t)4 * C_ * HW_ * 8 * 4);
    float*     agg   = (float*)alloc(NELEM * 4);               // [C,H,W,D]
    _Float16*  yb    = (_Float16*)alloc(NELEM * 2);            // [D,H,W,C]
    uint32_t*  bfrag = (uint32_t*)alloc((size_t)BFRAG_DW * 4);

    transpose_x_kernel<<<C_ * H_, 128, 0, stream>>>(x, xt);
    norm_guidance_kernel<<<(4 * C_ * HW_ + 255) / 256, 256, 0, stream>>>(g, kt);

    // specs = [(W,fwd), (W,rev), (H,fwd), (H,rev)]; 4 threads per row
    sga_scan_kernel<0, 0, 1><<<(C_ * H_ * 4 + 255) / 256, 256, 0, stream>>>(xt, kt, agg, 0);
    sga_scan_kernel<0, 1, 0><<<(C_ * H_ * 4 + 255) / 256, 256, 0, stream>>>(xt, kt, agg, 1);
    sga_scan_kernel<1, 0, 0><<<(C_ * W_ * 4 + 255) / 256, 256, 0, stream>>>(xt, kt, agg, 2);
    sga_scan_kernel<1, 1, 0><<<(C_ * W_ * 4 + 255) / 256, 256, 0, stream>>>(xt, kt, agg, 3);

    bn1_pack_kernel<<<(HW_ * 32 + 255) / 256, 256, 0, stream>>>(agg, s1, b1, yb);
    prep_bfrag_kernel<<<54, 32, 0, stream>>>(wr, bfrag);
    conv_wmma_kernel<<<D_ * H_, 256, 0, stream>>>(yb, bfrag, x, s2, b2, out);
}
